// Step_3_categories_1597727834238
// MI455X (gfx1250) — matrix operand order repaired and checked by reading, not compiled, not run on hardware
//
#include <hip/hip_runtime.h>
#include <hip/hip_bf16.h>

typedef __attribute__((ext_vector_type(16))) _Float16 v16h;
typedef __attribute__((ext_vector_type(8)))  _Float16 v8h;
typedef __attribute__((ext_vector_type(8)))  float    v8f;

#define B_   16
#define S_   128
#define D_   768
#define P_   1024
#define NCLS 121

// ---------------- weight convert + transpose: W[K,N] f32 -> Wt[Npad,K] f16 ----------------
__global__ void wt_transpose_f16(const float* __restrict__ W, _Float16* __restrict__ Wt,
                                 int K, int N, int Npad) {
    size_t idx = (size_t)blockIdx.x * blockDim.x + threadIdx.x;
    size_t total = (size_t)Npad * (size_t)K;
    if (idx >= total) return;
    int k = (int)(idx % (size_t)K);
    int n = (int)(idx / (size_t)K);
    float v = (n < N) ? W[(size_t)k * N + n] : 0.0f;
    Wt[idx] = (_Float16)v;
}

// ---------------- feature builder: X[p, 0:D]=aspect+opinion, [D:2D]=left, [2D:3D]=right ----
__global__ void feature_kernel(const float* __restrict__ emb,   // [B,S,D]
                               const int*   __restrict__ masks, // [B,S]
                               const int*   __restrict__ bidx,
                               const int*   __restrict__ aidx,
                               const int*   __restrict__ oidx,
                               _Float16*    __restrict__ X)     // [P, 3D] f16
{
    int p   = blockIdx.x;
    int tid = threadIdx.x;
    int bi = bidx[p], ai = aidx[p], oi = oidx[p];

    __shared__ int red[256];
    red[tid] = (tid < S_) ? masks[bi * S_ + tid] : 0;
    __syncthreads();
    for (int off = 128; off > 0; off >>= 1) {
        if (tid < off) red[tid] += red[tid + off];
        __syncthreads();
    }
    int li = red[0] - 1;   // last valid span index
    __syncthreads();

    bool both0 = (ai == 0) && (oi == 0);
    int  b     = (ai != 0) ? ai : oi;
    int  o_eff = (oi == 0) ? li : oi;

    const float* eb = emb + (size_t)bi * S_ * D_;
    _Float16* xp = X + (size_t)p * (3 * D_);

    for (int d = tid; d < D_; d += blockDim.x) {
        float left, right;
        if (both0) {
            float s = 0.0f, mx = -INFINITY;
            for (int j = 1; j < li; ++j) {
                float e = eb[(size_t)j * D_ + d];
                s += e; mx = fmaxf(mx, e);
            }
            float cnt = (float)(li - 1);
            if (cnt < 1.0f) cnt = 1.0f;
            left  = s / cnt;
            right = mx;
        } else {
            if (b == 1) {
                left = eb[d];                      // emb0
            } else {
                float mx = -INFINITY;
                for (int j = 1; j < b; ++j) mx = fmaxf(mx, eb[(size_t)j * D_ + d]);
                left = mx;
            }
            if (b == li) {
                right = eb[(size_t)li * D_ + d];   // embL
            } else {
                float mx = -INFINITY;
                for (int j = b; j < li; ++j) mx = fmaxf(mx, eb[(size_t)j * D_ + d]);
                right = mx;
            }
        }
        float srep = eb[(size_t)ai * D_ + d] + eb[(size_t)o_eff * D_ + d];
        xp[d]          = (_Float16)srep;
        xp[D_ + d]     = (_Float16)left;
        xp[2 * D_ + d] = (_Float16)right;
    }
}

// ---------------- WMMA fragment loaders (direct from global, L2-resident) ----------------
__device__ __forceinline__ v16h frag_cat(v8h lo, v8h hi) {
    return __builtin_shufflevector(lo, hi, 0, 1, 2, 3, 4, 5, 6, 7,
                                           8, 9, 10, 11, 12, 13, 14, 15);
}

// 16-bit A 16x32: lanes<16 -> row=lane, K {0..7, 16..23}; lanes>=16 -> row=lane-16, K {8..15, 24..31}
__device__ __forceinline__ v16h load_a_frag(const _Float16* __restrict__ A, int lda,
                                            int mBase, int k, int lane) {
    int half = lane >> 4;
    const _Float16* p = A + (size_t)(mBase + (lane & 15)) * lda + k + half * 8;
    v8h lo = *(const v8h*)(p);
    v8h hi = *(const v8h*)(p + 16);
    return frag_cat(lo, hi);
}

// 16-bit B 32x16 from Bt[N,K] row-major: lane = column (mod 16); lanes<16 K {0..7, 16..23}, lanes>=16 K {8..15, 24..31}
__device__ __forceinline__ v16h load_b_frag(const _Float16* __restrict__ Bt, int ldb,
                                            int nBase, int k, int lane) {
    const _Float16* p = Bt + (size_t)(nBase + (lane & 15)) * ldb + k + (lane >> 4) * 8;
    v8h lo = *(const v8h*)(p);
    v8h hi = *(const v8h*)(p + 16);
    return frag_cat(lo, hi);
}

// ---------------- WMMA GEMM: C[M,N] = act(A[M,K] @ Bt[N,K]^T + bias) ----------------
// 256 threads = 8 waves; waves 4(M) x 2(N); per-wave C tile 32x64 (2x4 WMMA tiles);
// block tile 128x128. 12 b128 loads per 8 WMMAs per k-step (1.5 loads/WMMA).
template <bool RELU, bool OUT16>
__global__ void gemm_wmma(const _Float16* __restrict__ A,
                          const _Float16* __restrict__ Bt,
                          const float*    __restrict__ bias, int nvalid,
                          void* __restrict__ Cout, int ldc,
                          int M, int N, int K)
{
    int lane = threadIdx.x & 31;
    int wave = threadIdx.x >> 5;
    int wm = wave & 3;
    int wn = wave >> 2;
    int m0 = blockIdx.x * 128 + wm * 32;
    int n0 = blockIdx.y * 128 + wn * 64;

    v8f c[2][4];
#pragma unroll
    for (int i = 0; i < 2; ++i)
#pragma unroll
        for (int j = 0; j < 4; ++j)
            c[i][j] = (v8f){};

    for (int k = 0; k < K; k += 32) {
        v16h a0 = load_a_frag(A, K, m0,      k, lane);
        v16h a1 = load_a_frag(A, K, m0 + 16, k, lane);
        v16h b0 = load_b_frag(Bt, K, n0,      k, lane);
        v16h b1 = load_b_frag(Bt, K, n0 + 16, k, lane);
        v16h b2 = load_b_frag(Bt, K, n0 + 32, k, lane);
        v16h b3 = load_b_frag(Bt, K, n0 + 48, k, lane);
        c[0][0] = __builtin_amdgcn_wmma_f32_16x16x32_f16(false, a0, false, b0, (short)0, c[0][0], false, false);
        c[0][1] = __builtin_amdgcn_wmma_f32_16x16x32_f16(false, a0, false, b1, (short)0, c[0][1], false, false);
        c[0][2] = __builtin_amdgcn_wmma_f32_16x16x32_f16(false, a0, false, b2, (short)0, c[0][2], false, false);
        c[0][3] = __builtin_amdgcn_wmma_f32_16x16x32_f16(false, a0, false, b3, (short)0, c[0][3], false, false);
        c[1][0] = __builtin_amdgcn_wmma_f32_16x16x32_f16(false, a1, false, b0, (short)0, c[1][0], false, false);
        c[1][1] = __builtin_amdgcn_wmma_f32_16x16x32_f16(false, a1, false, b1, (short)0, c[1][1], false, false);
        c[1][2] = __builtin_amdgcn_wmma_f32_16x16x32_f16(false, a1, false, b2, (short)0, c[1][2], false, false);
        c[1][3] = __builtin_amdgcn_wmma_f32_16x16x32_f16(false, a1, false, b3, (short)0, c[1][3], false, false);
    }

    // C/D layout: VGPR r, lanes<16 -> row r, col lane; lanes>=16 -> row r+8, col lane-16
    int rbase = (lane >> 4) * 8;
    int ccol  = lane & 15;

#pragma unroll
    for (int tm = 0; tm < 2; ++tm) {
#pragma unroll
        for (int tn = 0; tn < 4; ++tn) {
            int col = n0 + tn * 16 + ccol;
            if (col < nvalid) {
                float bv = bias[col];
#pragma unroll
                for (int r = 0; r < 8; ++r) {
                    int row = m0 + tm * 16 + rbase + r;
                    float v = c[tm][tn][r] + bv;
                    if (RELU) v = fmaxf(v, 0.0f);
                    if (OUT16) ((_Float16*)Cout)[(size_t)row * ldc + col] = (_Float16)v;
                    else       ((float*)Cout)[(size_t)row * ldc + col]    = v;
                }
            }
        }
    }
}

extern "C" void kernel_launch(void* const* d_in, const int* in_sizes, int n_in,
                              void* d_out, int out_size, void* d_ws, size_t ws_size,
                              hipStream_t stream)
{
    (void)in_sizes; (void)n_in; (void)out_size; (void)ws_size;

    const float* emb   = (const float*)d_in[0];
    const float* W1    = (const float*)d_in[1];
    const float* b1    = (const float*)d_in[2];
    const float* W2    = (const float*)d_in[3];
    const float* b2    = (const float*)d_in[4];
    const float* W3    = (const float*)d_in[5];
    const float* b3    = (const float*)d_in[6];
    const int*   masks = (const int*)d_in[7];
    const int*   bidx  = (const int*)d_in[8];
    const int*   aidx  = (const int*)d_in[9];
    const int*   oidx  = (const int*)d_in[10];
    float* out = (float*)d_out;

    char* ws = (char*)d_ws;
    size_t off = 0;
    auto alloc = [&](size_t elems) -> _Float16* {
        _Float16* p = (_Float16*)(ws + off);
        off += ((elems * sizeof(_Float16) + 255) & ~(size_t)255);
        return p;
    };
    _Float16* X   = alloc((size_t)P_ * 3 * D_);   // [1024,2304]
    _Float16* W1t = alloc((size_t)1536 * 2304);   // [N,K]
    _Float16* W2t = alloc((size_t)768 * 1536);
    _Float16* W3t = alloc((size_t)128 * 768);     // N padded 121->128 (zero rows)
    _Float16* H1  = alloc((size_t)P_ * 1536);
    _Float16* H2  = alloc((size_t)P_ * 768);

    {   // W1 [2304,1536] -> W1t [1536,2304]
        size_t tot = (size_t)1536 * 2304;
        wt_transpose_f16<<<(unsigned)((tot + 255) / 256), 256, 0, stream>>>(W1, W1t, 2304, 1536, 1536);
    }
    {   // W2 [1536,768] -> W2t [768,1536]
        size_t tot = (size_t)768 * 1536;
        wt_transpose_f16<<<(unsigned)((tot + 255) / 256), 256, 0, stream>>>(W2, W2t, 1536, 768, 768);
    }
    {   // W3 [768,121] -> W3t [128,768] (zero-padded)
        size_t tot = (size_t)128 * 768;
        wt_transpose_f16<<<(unsigned)((tot + 255) / 256), 256, 0, stream>>>(W3, W3t, 768, NCLS, 128);
    }

    feature_kernel<<<P_, 256, 0, stream>>>(emb, masks, bidx, aidx, oidx, X);

    // H1 = relu(X @ W1 + b1)       [1024,1536]
    gemm_wmma<true,  true ><<<dim3(P_ / 128, 1536 / 128), 256, 0, stream>>>(
        X, W1t, b1, 1536, (void*)H1, 1536, P_, 1536, 2304);
    // H2 = relu(H1 @ W2 + b2)      [1024,768]
    gemm_wmma<true,  true ><<<dim3(P_ / 128, 768 / 128), 256, 0, stream>>>(
        H1, W2t, b2, 768, (void*)H2, 768, P_, 768, 1536);
    // out = H2 @ W3 + b3           [1024,121] fp32
    gemm_wmma<false, false><<<dim3(P_ / 128, 128 / 128), 256, 0, stream>>>(
        H2, W3t, b3, NCLS, (void*)out, NCLS, P_, 128, 768);
}
